// GRAFTAdapter_40785009443412
// MI455X (gfx1250) — matrix operand order repaired and checked by reading, not compiled, not run on hardware
//
#include <hip/hip_runtime.h>
#include <hip/hip_bf16.h>

#define NNODE 4096
#define NEDGE 65536
#define DIN   128
#define DMODEL 256
#define NHEAD 4
#define NLAYER 3
#define DHEAD 64

typedef __attribute__((ext_vector_type(16))) _Float16 v16h;
typedef __attribute__((ext_vector_type(8)))  _Float16 v8h;
typedef __attribute__((ext_vector_type(8)))  float    v8f;
typedef __attribute__((ext_vector_type(4)))  float    v4f;
typedef __attribute__((ext_vector_type(4)))  int      i4;

// ---------------------------------------------------------------------------
// CDNA5 async global->LDS copy (ASYNCcnt path), with portable fallback
// ---------------------------------------------------------------------------
#if defined(__has_builtin)
#if __has_builtin(__builtin_amdgcn_global_load_async_to_lds_b128)
#define HAS_ASYNC_LDS 1
#endif
#endif
#ifndef HAS_ASYNC_LDS
#define HAS_ASYNC_LDS 0
#endif

typedef __attribute__((address_space(1))) i4 gi4;   // global int4
typedef __attribute__((address_space(3))) i4 li4;   // LDS int4

// copy 8 halfs (16 bytes) per lane from global to LDS
__device__ __forceinline__ void lds_cp16(_Float16* dst_lds, const _Float16* src_glb) {
#if HAS_ASYNC_LDS
    __builtin_amdgcn_global_load_async_to_lds_b128(
        (gi4*)(unsigned long long)src_glb,
        (li4*)(unsigned)(unsigned long long)dst_lds, 0, 0);
#else
    *(v8h*)dst_lds = *(const v8h*)src_glb;
#endif
}

__device__ __forceinline__ void lds_cp_wait() {
#if HAS_ASYNC_LDS
#if __has_builtin(__builtin_amdgcn_s_wait_asynccnt)
    __builtin_amdgcn_s_wait_asynccnt(0);
#else
    asm volatile("s_wait_asynccnt 0" ::: "memory");
#endif
#endif
}

// ---------------------------------------------------------------------------
// small utility kernels
// ---------------------------------------------------------------------------
__global__ void k_f32_to_f16(const float* __restrict__ src, _Float16* __restrict__ dst, int n) {
    int i = blockIdx.x * blockDim.x + threadIdx.x;
    if (i < n) dst[i] = (_Float16)src[i];
}

__global__ void k_deg_init(float* deg, int n) {
    int i = blockIdx.x * blockDim.x + threadIdx.x;
    if (i < n) deg[i] = 1.0f;   // self loop
}

__global__ void k_deg_accum(const int* __restrict__ col, float* deg, int e) {
    int i = blockIdx.x * blockDim.x + threadIdx.x;
    if (i < e) atomicAdd(&deg[col[i]], 1.0f);
}

__global__ void k_make_dinv(const float* __restrict__ deg, float* dinv, int n) {
    int i = blockIdx.x * blockDim.x + threadIdx.x;
    if (i < n) dinv[i] = rsqrtf(deg[i]);
}

__global__ void k_agg_self(const float* __restrict__ msg, const float* __restrict__ dinv,
                           float* __restrict__ agg, int total) {
    int i = blockIdx.x * blockDim.x + threadIdx.x;
    if (i >= total) return;
    int node = i >> 8;               // D == 256
    float s = dinv[node];
    agg[i] = s * s * msg[i];
}

__global__ void k_agg_edges(const int* __restrict__ row, const int* __restrict__ col,
                            const float* __restrict__ dinv, const float* __restrict__ msg,
                            float* __restrict__ agg, int e) {
    int idx = blockIdx.x * blockDim.x + threadIdx.x;
    int ed = idx >> 6;
    if (ed >= e) return;
    int dq = (idx & 63) * 4;
    int r = row[ed], c = col[ed];
    float nrm = dinv[r] * dinv[c];
    v4f mv = *(const v4f*)(msg + (size_t)r * DMODEL + dq);
    float* dst = agg + (size_t)c * DMODEL + dq;
    atomicAdd(dst + 0, nrm * mv[0]);
    atomicAdd(dst + 1, nrm * mv[1]);
    atomicAdd(dst + 2, nrm * mv[2]);
    atomicAdd(dst + 3, nrm * mv[3]);
}

__global__ void k_gcn_finalize(const float* __restrict__ agg, const float* __restrict__ b,
                               _Float16* __restrict__ outh, float* __restrict__ outf,
                               int total, int relu) {
    int i = blockIdx.x * blockDim.x + threadIdx.x;
    if (i >= total) return;
    float v = agg[i] + b[i & (DMODEL - 1)];
    if (relu) v = v > 0.f ? v : 0.f;
    outh[i] = (_Float16)v;
    if (outf) outf[i] = v;
}

__global__ void k_edge_bias(const int* __restrict__ row, const int* __restrict__ col,
                            const float* __restrict__ ew, const float* __restrict__ eb,
                            float* __restrict__ bias, int e) {
    int i = blockIdx.x * blockDim.x + threadIdx.x;
    if (i >= e) return;
    int r = row[i], c = col[i];
    float fr = (float)r, fc = (float)c;
#pragma unroll
    for (int h = 0; h < NHEAD; h++) {
        float z = fr * ew[h * 3 + 0] + fc * ew[h * 3 + 1] + ew[h * 3 + 2] + eb[h];
        float p = 1.0f / (1.0f + __expf(-z));
        bias[((size_t)h << 24) + ((size_t)r << 12) + (size_t)c] = p;
    }
}

__global__ __launch_bounds__(256) void k_layernorm(const float* __restrict__ in,
                                                   const float* __restrict__ g,
                                                   const float* __restrict__ b,
                                                   float* __restrict__ outf,
                                                   _Float16* __restrict__ outh, int nrows) {
    int wave = threadIdx.x >> 5, lane = threadIdx.x & 31;
    int rowid = blockIdx.x * 8 + wave;
    if (rowid >= nrows) return;
    const float* x = in + (size_t)rowid * DMODEL;
    float vals[8];
    float s = 0.f;
#pragma unroll
    for (int i = 0; i < 8; i++) { vals[i] = x[lane + 32 * i]; s += vals[i]; }
#pragma unroll
    for (int m = 16; m >= 1; m >>= 1) s += __shfl_xor(s, m, 32);
    float mean = s * (1.0f / DMODEL);
    float vs = 0.f;
#pragma unroll
    for (int i = 0; i < 8; i++) { float d = vals[i] - mean; vs += d * d; }
#pragma unroll
    for (int m = 16; m >= 1; m >>= 1) vs += __shfl_xor(vs, m, 32);
    float rs = rsqrtf(vs * (1.0f / DMODEL) + 1e-5f);
#pragma unroll
    for (int i = 0; i < 8; i++) {
        int d = lane + 32 * i;
        float o = (vals[i] - mean) * rs * g[d] + b[d];
        outf[(size_t)rowid * DMODEL + d] = o;
        outh[(size_t)rowid * DMODEL + d] = (_Float16)o;
    }
}

__global__ __launch_bounds__(256) void k_out_head(const float* __restrict__ x,
                                                  const float* __restrict__ w,
                                                  const float* __restrict__ b,
                                                  float* __restrict__ out, int nrows) {
    int wave = threadIdx.x >> 5, lane = threadIdx.x & 31;
    int rowid = blockIdx.x * 8 + wave;
    if (rowid >= nrows) return;
    const float* xr = x + (size_t)rowid * DMODEL;
    float s = 0.f;
#pragma unroll
    for (int i = 0; i < 8; i++) { int d = lane + 32 * i; s += xr[d] * w[d]; }
#pragma unroll
    for (int m = 16; m >= 1; m >>= 1) s += __shfl_xor(s, m, 32);
    if (lane == 0) out[rowid] = 1.0f / (1.0f + __expf(-(s + b[0])));
}

// ---------------------------------------------------------------------------
// WMMA GEMM: C[M,N] = A[M,K] (f16, lda) @ B[N,K]^T (f16, ldb)
//            (+ bias[n]) (+ resid[m,n]) (relu?) -> Cf (f32) and/or Ch (f16)
// Block: 64x64 tile, 4 waves.  B tile (shared by all 4 waves) is staged in
// LDS via double-buffered async global->LDS copies.
// ---------------------------------------------------------------------------
__global__ __launch_bounds__(128) void k_gemm_f16(
    const _Float16* __restrict__ A, int lda,
    const _Float16* __restrict__ B, int ldb,
    const float* __restrict__ bias,
    const float* __restrict__ resid, int ldr,
    float* __restrict__ Cf, int ldc,
    _Float16* __restrict__ Ch, int ldch,
    int K, int act) {
    __shared__ _Float16 ldsB[2][64 * 32];   // 4KB per buffer
    int tid = threadIdx.x;
    int wave = tid >> 5, lane = tid & 31;
    int lg = lane >> 4, ln = lane & 15;
    int m0 = blockIdx.x * 64 + wave * 16;
    int n0 = blockIdx.y * 64;

    // stage one 64x32 B tile (rows n0..n0+63, cols k0..k0+31) into ldsB[p]
    auto stageB = [&](int k0, int p) {
#pragma unroll
        for (int i = 0; i < 2; i++) {
            int chunk = tid + i * 128;      // 256 chunks of 8 halfs
            int r = chunk >> 2;
            int co = (chunk & 3) * 8;
            lds_cp16(&ldsB[p][r * 32 + co], B + (size_t)(n0 + r) * ldb + k0 + co);
        }
    };

    v8f acc[4];
#pragma unroll
    for (int j = 0; j < 4; j++)
#pragma unroll
        for (int v = 0; v < 8; v++) acc[j][v] = 0.f;

    stageB(0, 0);
    for (int k0 = 0; k0 < K; k0 += 32) {
        int p = (k0 >> 5) & 1;
        lds_cp_wait();
        __syncthreads();
        if (k0 + 32 < K) stageB(k0 + 32, p ^ 1);

        // A fragment (16x32): lane = m + 16*((k>>3)&1), halfs (k&7)+8*(k>>4)
        int arow = m0 + ln;
        int kb = k0 + 8 * lg;
        v8h a0 = *(const v8h*)(A + (size_t)arow * lda + kb);
        v8h a1 = *(const v8h*)(A + (size_t)arow * lda + kb + 16);
        v16h a;
#pragma unroll
        for (int i = 0; i < 8; i++) { a[i] = a0[i]; a[8 + i] = a1[i]; }
#pragma unroll
        for (int j = 0; j < 4; j++) {
            // B fragment from LDS: lane = n + 16*((k>>3)&1), halfs (k&7)+8*(k>>4)
            const _Float16* bp = &ldsB[p][(16 * j + ln) * 32 + 8 * lg];
            v8h b0 = *(const v8h*)(bp);
            v8h b1 = *(const v8h*)(bp + 16);
            v16h bb;
#pragma unroll
            for (int i = 0; i < 8; i++) { bb[i] = b0[i]; bb[8 + i] = b1[i]; }
            acc[j] = __builtin_amdgcn_wmma_f32_16x16x32_f16(
                false, a, false, bb, (short)0, acc[j], false, false);
        }
    }
    // epilogue: C layout lane = n + 16*(m>>3), vgpr = m&7
#pragma unroll
    for (int j = 0; j < 4; j++) {
        int n = n0 + 16 * j + ln;
        float bv = bias ? bias[n] : 0.f;
#pragma unroll
        for (int v = 0; v < 8; v++) {
            int m = m0 + 8 * lg + v;
            float c = acc[j][v] + bv;
            if (resid) c += resid[(size_t)m * ldr + n];
            if (act == 1) c = c > 0.f ? c : 0.f;
            if (Cf) Cf[(size_t)m * ldc + n] = c;
            if (Ch) Ch[(size_t)m * ldch + n] = (_Float16)c;
        }
    }
}

// ---------------------------------------------------------------------------
// Flash attention: qkv is f16 [N, 768]; bias f32 [H,N,N]; out f16 [N,256].
// Block = 64 query rows x 1 head; 4 waves.  K/V tiles (shared by all waves)
// staged in LDS via double-buffered async copies; online softmax in regs;
// P goes through a per-wave LDS transpose to feed the PV WMMA.
// ---------------------------------------------------------------------------
#define PSTRIDE 72

__global__ __launch_bounds__(128) void k_attention(
    const _Float16* __restrict__ qkv,
    const float* __restrict__ bias,
    _Float16* __restrict__ attn_h) {
    __shared__ _Float16 ldsK[2][64 * 64];   // 8KB each
    __shared__ _Float16 ldsV[2][64 * 64];
    __shared__ _Float16 ldsP[64 * PSTRIDE];
    int tid = threadIdx.x;
    int wave = tid >> 5, lane = tid & 31;
    int lg = lane >> 4, ln = lane & 15;
    int h = blockIdx.y;
    int mw = blockIdx.x * 64 + wave * 16;
    const int QO = h * DHEAD, KO = DMODEL + h * DHEAD, VO = 2 * DMODEL + h * DHEAD;
    const float scale = 0.125f;             // 1/sqrt(64)

    // stage K and V tiles for key rows t0..t0+63 into buffers [p]
    auto stage = [&](int t0, int p) {
#pragma unroll
        for (int i = 0; i < 4; i++) {
            int chunk = tid + i * 128;      // 512 chunks of 8 halfs
            int r = chunk >> 3;
            int co = (chunk & 7) * 8;
            const _Float16* base = qkv + (size_t)(t0 + r) * 768;
            lds_cp16(&ldsK[p][r * 64 + co], base + KO + co);
            lds_cp16(&ldsV[p][r * 64 + co], base + VO + co);
        }
    };

    // preload Q fragments (A layout), straight b128 loads from f16 qkv
    v16h aq[2];
#pragma unroll
    for (int s = 0; s < 2; s++) {
        int row = mw + ln;
        int kb = 32 * s + 8 * lg;
        const _Float16* p = qkv + (size_t)row * 768 + QO + kb;
        v8h q0 = *(const v8h*)(p);
        v8h q1 = *(const v8h*)(p + 16);
        v16h a;
#pragma unroll
        for (int i = 0; i < 8; i++) { a[i] = q0[i]; a[8 + i] = q1[i]; }
        aq[s] = a;
    }

    float Mv[8], Lv[8];
    v8f o[4];
#pragma unroll
    for (int v = 0; v < 8; v++) { Mv[v] = -3.0e38f; Lv[v] = 0.f; }
#pragma unroll
    for (int j = 0; j < 4; j++)
#pragma unroll
        for (int v = 0; v < 8; v++) o[j][v] = 0.f;

    stage(0, 0);
    for (int t0 = 0; t0 < NNODE; t0 += 64) {
        int p = (t0 >> 6) & 1;
        lds_cp_wait();
        __syncthreads();
        if (t0 + 64 < NNODE) {
            stage(t0 + 64, p ^ 1);
            // speculative prefetch of next tile's bias rows (HBM-resident)
            __builtin_prefetch((const void*)(bias + ((size_t)h << 24) +
                                             ((size_t)(mw + 8 * lg) << 12) +
                                             (size_t)(t0 + 64) + ln), 0, 1);
        }
        // ---- S = Q K^T ---------------------------------------------------
        v8f sacc[4];
#pragma unroll
        for (int j = 0; j < 4; j++)
#pragma unroll
            for (int v = 0; v < 8; v++) sacc[j][v] = 0.f;
#pragma unroll
        for (int s = 0; s < 2; s++) {
#pragma unroll
            for (int j = 0; j < 4; j++) {
                const _Float16* bp = &ldsK[p][(16 * j + ln) * 64 + 32 * s + 8 * lg];
                v8h b0 = *(const v8h*)(bp);
                v8h b1 = *(const v8h*)(bp + 16);
                v16h bk;
#pragma unroll
                for (int i = 0; i < 8; i++) { bk[i] = b0[i]; bk[8 + i] = b1[i]; }
                sacc[j] = __builtin_amdgcn_wmma_f32_16x16x32_f16(
                    false, aq[s], false, bk, (short)0, sacc[j], false, false);
            }
        }
        // ---- scale + bias ------------------------------------------------
#pragma unroll
        for (int j = 0; j < 4; j++) {
            int n = t0 + 16 * j + ln;
#pragma unroll
            for (int v = 0; v < 8; v++) {
                int m = mw + 8 * lg + v;
                sacc[j][v] = sacc[j][v] * scale +
                             bias[((size_t)h << 24) + ((size_t)m << 12) + (size_t)n];
            }
        }
        // ---- online softmax (row lives in a 16-lane half-wave group) ----
#pragma unroll
        for (int v = 0; v < 8; v++) {
            float rmax = sacc[0][v];
#pragma unroll
            for (int j = 1; j < 4; j++) rmax = fmaxf(rmax, sacc[j][v]);
#pragma unroll
            for (int msk = 8; msk >= 1; msk >>= 1) rmax = fmaxf(rmax, __shfl_xor(rmax, msk, 32));
            float nm = fmaxf(Mv[v], rmax);
            float corr = __expf(Mv[v] - nm);
            float rsum = 0.f;
#pragma unroll
            for (int j = 0; j < 4; j++) {
                float pv = __expf(sacc[j][v] - nm);
                sacc[j][v] = pv;
                rsum += pv;
            }
#pragma unroll
            for (int msk = 8; msk >= 1; msk >>= 1) rsum += __shfl_xor(rsum, msk, 32);
            Lv[v] = Lv[v] * corr + rsum;
#pragma unroll
            for (int j = 0; j < 4; j++) o[j][v] *= corr;
            Mv[v] = nm;
        }
        // ---- P: C layout -> LDS -> A layout (intra-wave, LDS in-order) --
#pragma unroll
        for (int j = 0; j < 4; j++)
#pragma unroll
            for (int v = 0; v < 8; v++)
                ldsP[(wave * 16 + 8 * lg + v) * PSTRIDE + 16 * j + ln] =
                    (_Float16)sacc[j][v];
#pragma unroll
        for (int s = 0; s < 2; s++) {
            int prow = wave * 16 + ln;
            int kb = 32 * s + 16 * lg;
            v8h p0 = *(const v8h*)(&ldsP[prow * PSTRIDE + kb]);
            v8h p1 = *(const v8h*)(&ldsP[prow * PSTRIDE + kb + 8]);
            v16h ap;
#pragma unroll
            for (int i = 0; i < 8; i++) { ap[i] = p0[i]; ap[8 + i] = p1[i]; }
#pragma unroll
            for (int j = 0; j < 4; j++) {
                // V B-fragment: lane = out dim, halfs walk 16 key rows in LDS
                v16h bv;
#pragma unroll
                for (int t = 0; t < 16; t++)
                    bv[t] = ldsV[p][(32 * s + 16 * lg + t) * 64 + 16 * j + ln];
                o[j] = __builtin_amdgcn_wmma_f32_16x16x32_f16(
                    false, ap, false, bv, (short)0, o[j], false, false);
            }
        }
    }
    // ---- normalize + store f16 ------------------------------------------
#pragma unroll
    for (int j = 0; j < 4; j++) {
#pragma unroll
        for (int v = 0; v < 8; v++) {
            int m = mw + 8 * lg + v;
            float val = o[j][v] / Lv[v];
            attn_h[(size_t)m * DMODEL + h * DHEAD + 16 * j + ln] = (_Float16)val;
        }
    }
}

// ---------------------------------------------------------------------------
// host-side orchestration
// ---------------------------------------------------------------------------
extern "C" void kernel_launch(void* const* d_in, const int* in_sizes, int n_in,
                              void* d_out, int out_size, void* d_ws, size_t ws_size,
                              hipStream_t stream) {
    (void)in_sizes; (void)n_in; (void)out_size; (void)ws_size;
    const float* nf      = (const float*)d_in[0];
    const int*   eidx    = (const int*)d_in[1];
    const float* gcn1_w  = (const float*)d_in[2];
    const float* gcn1_b  = (const float*)d_in[3];
    const float* gcn2_w  = (const float*)d_in[4];
    const float* gcn2_b  = (const float*)d_in[5];
    const float* edge_w  = (const float*)d_in[6];
    const float* edge_b  = (const float*)d_in[7];
    const float* proj_w  = (const float*)d_in[8];
    const float* proj_b  = (const float*)d_in[9];
    const float* wqkv    = (const float*)d_in[10];
    const float* bqkv    = (const float*)d_in[11];
    const float* wo      = (const float*)d_in[12];
    const float* bo      = (const float*)d_in[13];
    const float* ffn_w1  = (const float*)d_in[14];
    const float* ffn_b1  = (const float*)d_in[15];
    const float* ffn_w2  = (const float*)d_in[16];
    const float* ffn_b2  = (const float*)d_in[17];
    const float* ln1_g   = (const float*)d_in[18];
    const float* ln1_b   = (const float*)d_in[19];
    const float* ln2_g   = (const float*)d_in[20];
    const float* ln2_b   = (const float*)d_in[21];
    const float* out_w   = (const float*)d_in[22];
    const float* out_b   = (const float*)d_in[23];
    float* outp = (float*)d_out;
    const int* erow = eidx;
    const int* ecol = eidx + NEDGE;

    char* wsp = (char*)d_ws;
    size_t off = 0;
    auto carve = [&](size_t bytes) {
        void* p = wsp + off;
        off = (off + bytes + 255) & ~(size_t)255;
        return p;
    };
    float*     biasT  = (float*)carve((size_t)NHEAD * NNODE * NNODE * 4);   // 256 MB
    _Float16*  nf_h   = (_Float16*)carve((size_t)NNODE * DIN * 2);
    _Float16*  g1w_h  = (_Float16*)carve((size_t)DMODEL * DIN * 2);
    _Float16*  g2w_h  = (_Float16*)carve((size_t)DMODEL * DMODEL * 2);
    _Float16*  pjw_h  = (_Float16*)carve((size_t)DMODEL * (DIN + DMODEL) * 2);
    _Float16*  wqkv_h = (_Float16*)carve((size_t)NLAYER * 3 * DMODEL * DMODEL * 2);
    _Float16*  wo_h   = (_Float16*)carve((size_t)NLAYER * DMODEL * DMODEL * 2);
    _Float16*  w1_h   = (_Float16*)carve((size_t)NLAYER * 4 * DMODEL * DMODEL * 2);
    _Float16*  w2_h   = (_Float16*)carve((size_t)NLAYER * DMODEL * 4 * DMODEL * 2);
    float*     deg    = (float*)carve((size_t)NNODE * 4);
    float*     dinv   = (float*)carve((size_t)NNODE * 4);
    float*     msg    = (float*)carve((size_t)NNODE * DMODEL * 4);
    float*     agg    = (float*)carve((size_t)NNODE * DMODEL * 4);
    _Float16*  h_h    = (_Float16*)carve((size_t)NNODE * DMODEL * 2);
    _Float16*  g_h    = (_Float16*)carve((size_t)NNODE * DMODEL * 2);
    float*     xbuf   = (float*)carve((size_t)NNODE * DMODEL * 4);
    _Float16*  x_h    = (_Float16*)carve((size_t)NNODE * DMODEL * 2);
    _Float16*  qkv_h  = (_Float16*)carve((size_t)NNODE * 3 * DMODEL * 2);
    _Float16*  attn_h = (_Float16*)carve((size_t)NNODE * DMODEL * 2);
    float*     t1     = (float*)carve((size_t)NNODE * DMODEL * 4);
    _Float16*  f1_h   = (_Float16*)carve((size_t)NNODE * 4 * DMODEL * 2);

    auto cvt = [&](const float* s, _Float16* d, int n) {
        k_f32_to_f16<<<(n + 255) / 256, 256, 0, stream>>>(s, d, n);
    };
    cvt(nf, nf_h, NNODE * DIN);
    cvt(gcn1_w, g1w_h, DMODEL * DIN);
    cvt(gcn2_w, g2w_h, DMODEL * DMODEL);
    cvt(proj_w, pjw_h, DMODEL * (DIN + DMODEL));
    cvt(wqkv, wqkv_h, NLAYER * 3 * DMODEL * DMODEL);
    cvt(wo, wo_h, NLAYER * DMODEL * DMODEL);
    cvt(ffn_w1, w1_h, NLAYER * 4 * DMODEL * DMODEL);
    cvt(ffn_w2, w2_h, NLAYER * DMODEL * 4 * DMODEL);

    k_deg_init<<<NNODE / 256, 256, 0, stream>>>(deg, NNODE);
    k_deg_accum<<<NEDGE / 256, 256, 0, stream>>>(ecol, deg, NEDGE);
    k_make_dinv<<<NNODE / 256, 256, 0, stream>>>(deg, dinv, NNODE);

    (void)hipMemsetAsync(biasT, 0, (size_t)NHEAD * NNODE * NNODE * 4, stream);
    k_edge_bias<<<NEDGE / 256, 256, 0, stream>>>(erow, ecol, edge_w, edge_b, biasT, NEDGE);

    dim3 blk(128);
    dim3 grd256(NNODE / 64, DMODEL / 64);
    dim3 grd768(NNODE / 64, 3 * DMODEL / 64);
    dim3 grd1024(NNODE / 64, 4 * DMODEL / 64);

    // GCN layer 1
    k_gemm_f16<<<grd256, blk, 0, stream>>>(nf_h, DIN, g1w_h, DIN, nullptr,
                                           nullptr, 0, msg, DMODEL, nullptr, 0, DIN, 0);
    k_agg_self<<<NNODE * DMODEL / 256, 256, 0, stream>>>(msg, dinv, agg, NNODE * DMODEL);
    k_agg_edges<<<NEDGE * 64 / 256, 256, 0, stream>>>(erow, ecol, dinv, msg, agg, NEDGE);
    k_gcn_finalize<<<NNODE * DMODEL / 256, 256, 0, stream>>>(agg, gcn1_b, h_h, nullptr,
                                                             NNODE * DMODEL, 1);
    // GCN layer 2
    k_gemm_f16<<<grd256, blk, 0, stream>>>(h_h, DMODEL, g2w_h, DMODEL, nullptr,
                                           nullptr, 0, msg, DMODEL, nullptr, 0, DMODEL, 0);
    k_agg_self<<<NNODE * DMODEL / 256, 256, 0, stream>>>(msg, dinv, agg, NNODE * DMODEL);
    k_agg_edges<<<NEDGE * 64 / 256, 256, 0, stream>>>(erow, ecol, dinv, msg, agg, NEDGE);
    k_gcn_finalize<<<NNODE * DMODEL / 256, 256, 0, stream>>>(agg, gcn2_b, g_h, nullptr,
                                                             NNODE * DMODEL, 0);
    // projection (K-split over [nf | g])
    k_gemm_f16<<<grd256, blk, 0, stream>>>(nf_h, DIN, pjw_h, DIN + DMODEL, nullptr,
                                           nullptr, 0, xbuf, DMODEL, nullptr, 0, DIN, 0);
    k_gemm_f16<<<grd256, blk, 0, stream>>>(g_h, DMODEL, pjw_h + DIN, DIN + DMODEL, proj_b,
                                           xbuf, DMODEL, xbuf, DMODEL, x_h, DMODEL, DMODEL, 0);

    for (int l = 0; l < NLAYER; l++) {
        const _Float16* wq  = wqkv_h + (size_t)l * 3 * DMODEL * DMODEL;
        const _Float16* wol = wo_h + (size_t)l * DMODEL * DMODEL;
        const _Float16* w1l = w1_h + (size_t)l * 4 * DMODEL * DMODEL;
        const _Float16* w2l = w2_h + (size_t)l * DMODEL * 4 * DMODEL;
        // qkv (f16 output only — consumed directly by flash attention)
        k_gemm_f16<<<grd768, blk, 0, stream>>>(x_h, DMODEL, wq, DMODEL, bqkv + l * 3 * DMODEL,
                                               nullptr, 0, nullptr, 0, qkv_h, 3 * DMODEL,
                                               DMODEL, 0);
        k_attention<<<dim3(NNODE / 64, NHEAD), blk, 0, stream>>>(qkv_h, biasT, attn_h);
        k_gemm_f16<<<grd256, blk, 0, stream>>>(attn_h, DMODEL, wol, DMODEL, bo + l * DMODEL,
                                               xbuf, DMODEL, t1, DMODEL, nullptr, 0, DMODEL, 0);
        k_layernorm<<<NNODE / 8, 256, 0, stream>>>(t1, ln1_g + l * DMODEL, ln1_b + l * DMODEL,
                                                   xbuf, x_h, NNODE);
        k_gemm_f16<<<grd1024, blk, 0, stream>>>(x_h, DMODEL, w1l, DMODEL,
                                                ffn_b1 + l * 4 * DMODEL, nullptr, 0,
                                                nullptr, 0, f1_h, 4 * DMODEL, DMODEL, 1);
        k_gemm_f16<<<grd256, blk, 0, stream>>>(f1_h, 4 * DMODEL, w2l, 4 * DMODEL,
                                               ffn_b2 + l * DMODEL, xbuf, DMODEL,
                                               t1, DMODEL, nullptr, 0, 4 * DMODEL, 0);
        k_layernorm<<<NNODE / 8, 256, 0, stream>>>(t1, ln2_g + l * DMODEL, ln2_b + l * DMODEL,
                                                   xbuf, x_h, NNODE);
    }
    k_out_head<<<NNODE / 8, 256, 0, stream>>>(xbuf, out_w, out_b, outp, NNODE);
}
